// CausalMultiHeadAttention_70514773065796
// MI455X (gfx1250) — matrix operand order repaired and checked by reading, not compiled, not run on hardware
//
#include <hip/hip_runtime.h>

typedef __attribute__((ext_vector_type(16))) _Float16 v16h;
typedef __attribute__((ext_vector_type(8)))  _Float16 v8h;
typedef __attribute__((ext_vector_type(4)))  _Float16 v4h;
typedef __attribute__((ext_vector_type(8)))  float    v8f;
typedef __attribute__((ext_vector_type(4)))  unsigned int v4u;
typedef __attribute__((ext_vector_type(8)))  int      v8i;
typedef __attribute__((ext_vector_type(4)))  int      v4i;

#define D_MODEL   1024
#define NUM_HEADS 16
#define HEAD_DIM  64
#define BATCH     4
#define SEQ       2048
#define TOKENS    (BATCH * SEQ)

#if __has_builtin(__builtin_amdgcn_tensor_load_to_lds) && \
    __has_builtin(__builtin_amdgcn_s_wait_tensorcnt)
#define HAVE_TDM 1
#else
#define HAVE_TDM 0
#endif

// LDS tile for the B (weight) operand: 64 rows x 128 halves, padded to
// 136 halves (272 B = 68 DWORDs) of pitch so fragment reads spread banks.
#define BK        128
#define B_PITCH   136

union Frag16 { v16h v; v8h h[2]; };

static __device__ inline v8f zero8() {
    v8f z;
#pragma unroll
    for (int i = 0; i < 8; ++i) z[i] = 0.0f;
    return z;
}

static __device__ inline v8f wmma16(const Frag16& a, const Frag16& b, v8f c) {
    // D = A(16x32 f16) * B(32x16 f16) + C(16x16 f32)
    return __builtin_amdgcn_wmma_f32_16x16x32_f16(
        false, a.v, false, b.v, (short)0, c, false, false);
}

// A fragment (16x32 f16): lane holds row m0+lr; halves K = (i/8)*16 + hi*8 + i%8
static __device__ inline void loadA(Frag16& a, const _Float16* row, int kk, int hi) {
    a.h[0] = *(const v8h*)(row + kk + hi * 8);
    a.h[1] = *(const v8h*)(row + kk + 16 + hi * 8);
}
// B fragment (32x16 f16) from LDS tile: lane holds col (tile row); K = (i/8)*16 + hi*8 + i%8
static __device__ inline void loadB_lds(Frag16& b, const _Float16* rowp, int kk, int hi) {
    b.h[0] = *(const v8h*)(rowp + kk + hi * 8);
    b.h[1] = *(const v8h*)(rowp + kk + 16 + hi * 8);
}

#if HAVE_TDM
// Issue a TDM DMA: 2D tile (64 rows x 128 f16) from global (row pitch 1024
// halves) into LDS at byte offset lds_off, LDS row pitch padded by 4 DWORDs
// per 64 DWORDs (-> 272 B). D# per CDNA5 ISA.
static __device__ inline void tdm_load_b_tile(const _Float16* gsrc, unsigned lds_off) {
    unsigned long long ga = (unsigned long long)(uintptr_t)gsrc;
    v4u g0;
    g0[0] = 1u;                                   // count=1, user descriptor
    g0[1] = lds_off;                              // lds_addr (bytes)
    g0[2] = (unsigned)(ga & 0xFFFFFFFFu);         // global_addr[31:0]
    g0[3] = (unsigned)((ga >> 32) & 0x01FFFFFFu)  // global_addr[56:32]
            | 0x80000000u;                        // type=2 ("image")
    v8i g1;
    // data_size=2B (code1), pad_enable=1, pad_interval=code5 (64 DW),
    // pad_amount=code3 (4 DW)
    g1[0] = (int)((1u << 16) | (1u << 20) | (5u << 22) | (3u << 25));
    g1[1] = (int)0;                               // dim0[15:0] in [31:16] (dim0=1M)
    g1[2] = (int)0x00000010;                      // dim0[31:16]=0x10; dim1[15:0]=0
    g1[3] = (int)(0x00000010 | (BK << 16));       // dim1[31:16]=0x10; tile_dim0=128
    g1[4] = (int)64;                              // tile_dim1=64; tile_dim2=0
    g1[5] = (int)D_MODEL;                         // tensor_dim0_stride = 1024 elems
    g1[6] = 0;
    g1[7] = 0;
    v4i g2 = {0, 0, 0, 0};
    v4i g3 = {0, 0, 0, 0};
#if __clang_major__ >= 23
    v8i g4 = {0, 0, 0, 0, 0, 0, 0, 0};
    __builtin_amdgcn_tensor_load_to_lds(g0, g1, g2, g3, g4, 0);
#else
    __builtin_amdgcn_tensor_load_to_lds(g0, g1, g2, g3, 0);
#endif
}
#endif

// ---------------------------------------------------------------------------
// fp32 -> f16 conversion (vectorized x4)
// ---------------------------------------------------------------------------
__global__ void cvt_f32_f16(const float* __restrict__ src,
                            _Float16* __restrict__ dst, int n4) {
    int i = blockIdx.x * blockDim.x + threadIdx.x;
    if (i < n4) {
        float4 f = ((const float4*)src)[i];
        v4h o;
        o[0] = (_Float16)f.x; o[1] = (_Float16)f.y;
        o[2] = (_Float16)f.z; o[3] = (_Float16)f.w;
        ((v4h*)dst)[i] = o;
    }
}

// ---------------------------------------------------------------------------
// Shared GEMM body: block = 128 rows x 64 cols, 8 waves, each wave a 32x32
// tile (2x2 register block). B tile staged in LDS by the TDM (double buffer).
// ---------------------------------------------------------------------------
#define GEMM_PROLOGUE()                                                        \
    const int wave = threadIdx.x >> 5;                                         \
    const int lane = threadIdx.x & 31;                                         \
    const int lr   = lane & 15;                                                \
    const int hi   = lane >> 4;                                                \
    const int wm   = wave & 3;                                                 \
    const int wn   = wave >> 2;                                                \
    const int m0    = blockIdx.x * 128 + wm * 32;                              \
    const int n0blk = blockIdx.y * 64;                                         \
    const int n0    = n0blk + wn * 32;

#if HAVE_TDM
#define GEMM_MAIN(Abase, Wbase)                                                \
    __shared__ __align__(16) _Float16 ldsB[2][64][B_PITCH];                    \
    const _Float16* arow0 = (Abase) + (m0 + lr) * D_MODEL;                     \
    const _Float16* arow1 = arow0 + 16 * D_MODEL;                              \
    v8f c00 = zero8(), c01 = zero8(), c10 = zero8(), c11 = zero8();            \
    if (wave == 0)                                                             \
        tdm_load_b_tile((Wbase) + n0blk * D_MODEL,                             \
                        (unsigned)(uintptr_t)&ldsB[0][0][0]);                  \
    int cur = 0;                                                               \
    for (int kc = 0; kc < D_MODEL; kc += BK, cur ^= 1) {                       \
        if (wave == 0) {                                                       \
            if (kc + BK < D_MODEL) {                                           \
                tdm_load_b_tile((Wbase) + n0blk * D_MODEL + kc + BK,           \
                                (unsigned)(uintptr_t)&ldsB[cur ^ 1][0][0]);    \
                __builtin_amdgcn_s_wait_tensorcnt(1);                          \
            } else {                                                           \
                __builtin_amdgcn_s_wait_tensorcnt(0);                          \
            }                                                                  \
        }                                                                      \
        __syncthreads();                                                       \
        const _Float16* bp0 = &ldsB[cur][wn * 32 + lr][0];                     \
        const _Float16* bp1 = bp0 + 16 * B_PITCH;                              \
        _Pragma("unroll")                                                      \
        for (int kk = 0; kk < BK; kk += 32) {                                  \
            Frag16 a0, a1, b0, b1;                                             \
            loadA(a0, arow0, kc + kk, hi);                                     \
            loadA(a1, arow1, kc + kk, hi);                                     \
            loadB_lds(b0, bp0, kk, hi);                                        \
            loadB_lds(b1, bp1, kk, hi);                                        \
            c00 = wmma16(a0, b0, c00);                                         \
            c01 = wmma16(a0, b1, c01);                                         \
            c10 = wmma16(a1, b0, c10);                                         \
            c11 = wmma16(a1, b1, c11);                                         \
        }                                                                      \
        __syncthreads();                                                       \
    }
#else
#define GEMM_MAIN(Abase, Wbase)                                                \
    __shared__ __align__(16) _Float16 ldsB[2][64][B_PITCH];                    \
    const _Float16* arow0 = (Abase) + (m0 + lr) * D_MODEL;                     \
    const _Float16* arow1 = arow0 + 16 * D_MODEL;                              \
    v8f c00 = zero8(), c01 = zero8(), c10 = zero8(), c11 = zero8();            \
    int cur = 0;                                                               \
    for (int kc = 0; kc < D_MODEL; kc += BK, cur ^= 1) {                       \
        /* cooperative copy: 64 rows x 16 chunks of 8 halves = 1024 chunks */  \
        for (int idx = threadIdx.x; idx < 64 * 16; idx += 256) {               \
            int r = idx >> 4, cch = idx & 15;                                  \
            *(v8h*)&ldsB[cur][r][cch * 8] =                                    \
                *(const v8h*)((Wbase) + (n0blk + r) * D_MODEL + kc + cch * 8); \
        }                                                                      \
        __syncthreads();                                                       \
        const _Float16* bp0 = &ldsB[cur][wn * 32 + lr][0];                     \
        const _Float16* bp1 = bp0 + 16 * B_PITCH;                              \
        _Pragma("unroll")                                                      \
        for (int kk = 0; kk < BK; kk += 32) {                                  \
            Frag16 a0, a1, b0, b1;                                             \
            loadA(a0, arow0, kc + kk, hi);                                     \
            loadA(a1, arow1, kc + kk, hi);                                     \
            loadB_lds(b0, bp0, kk, hi);                                        \
            loadB_lds(b1, bp1, kk, hi);                                        \
            c00 = wmma16(a0, b0, c00);                                         \
            c01 = wmma16(a0, b1, c01);                                         \
            c10 = wmma16(a1, b0, c10);                                         \
            c11 = wmma16(a1, b1, c11);                                         \
        }                                                                      \
        __syncthreads();                                                       \
    }
#endif

// ---------------------------------------------------------------------------
// QKV projection epilogue helpers
// ---------------------------------------------------------------------------
static __device__ inline void store_qkv_tile(
        const v8f& c, int m0s, int col, int which,
        _Float16* __restrict__ Q, _Float16* __restrict__ K,
        _Float16* __restrict__ VT, int hi) {
    const int bb = m0s >> 11;     // batch (tiles never cross batch boundary)
    const int s0 = m0s & 2047;
    const int h  = col >> 6;
    const int dh = col & 63;
    if (which == 2) {
        // V^T[b][h][dh][s]; lane's 8 values contiguous in s -> one 16B store
        _Float16* dst = VT + ((bb * NUM_HEADS + h) * HEAD_DIM + dh) * SEQ
                           + s0 + hi * 8;
        v8h o;
#pragma unroll
        for (int v = 0; v < 8; ++v) o[v] = (_Float16)c[v];
        *(v8h*)dst = o;
    } else {
        const float scale = (which == 0) ? 0.125f : 1.0f; // 1/sqrt(64)
        _Float16* dst = ((which == 0) ? Q : K)
                        + ((bb * NUM_HEADS + h) * SEQ) * HEAD_DIM + dh;
#pragma unroll
        for (int v = 0; v < 8; ++v) {
            int s = s0 + hi * 8 + v;
            dst[s * HEAD_DIM] = (_Float16)(c[v] * scale);
        }
    }
}

__global__ __launch_bounds__(256)
void qkv_kernel(const _Float16* __restrict__ xh,
                const _Float16* __restrict__ wq,
                const _Float16* __restrict__ wk,
                const _Float16* __restrict__ wv,
                _Float16* __restrict__ Q,
                _Float16* __restrict__ K,
                _Float16* __restrict__ VT) {
    GEMM_PROLOGUE();
    const int which = blockIdx.z;
    const _Float16* W = (which == 0) ? wq : (which == 1) ? wk : wv;
    GEMM_MAIN(xh, W);

    store_qkv_tile(c00, m0,      n0 + lr,      which, Q, K, VT, hi);
    store_qkv_tile(c01, m0,      n0 + 16 + lr, which, Q, K, VT, hi);
    store_qkv_tile(c10, m0 + 16, n0 + lr,      which, Q, K, VT, hi);
    store_qkv_tile(c11, m0 + 16, n0 + 16 + lr, which, Q, K, VT, hi);
}

// ---------------------------------------------------------------------------
// Output projection: out[t,n] = sum_k A[t,k] * Wo[n,k]  (fp32 result)
// ---------------------------------------------------------------------------
__global__ __launch_bounds__(256)
void oproj_kernel(const _Float16* __restrict__ A,
                  const _Float16* __restrict__ W,
                  float* __restrict__ out) {
    GEMM_PROLOGUE();
    GEMM_MAIN(A, W);

    float* orow0 = out + m0 * D_MODEL + n0 + lr;
    float* orow1 = orow0 + 16 * D_MODEL;
#pragma unroll
    for (int v = 0; v < 8; ++v) {
        const int r = (v + 8 * hi) * D_MODEL;
        orow0[r]      = c00[v];
        orow0[r + 16] = c01[v];
        orow1[r]      = c10[v];
        orow1[r + 16] = c11[v];
    }
}

// ---------------------------------------------------------------------------
// Flash-style causal attention. One wave per 16-row q-tile, 32 keys/step.
// 128 threads = 4 waves per block.
// ---------------------------------------------------------------------------
__global__ __launch_bounds__(128)
void attn_kernel(const _Float16* __restrict__ Q,
                 const _Float16* __restrict__ K,
                 const _Float16* __restrict__ VT,
                 _Float16* __restrict__ AO) {
    __shared__ __align__(16) _Float16 lds[4][16][40]; // 40-half row pitch: bank-spread

    const int wave = threadIdx.x >> 5;
    const int lane = threadIdx.x & 31;
    const int lr   = lane & 15;
    const int hi   = lane >> 4;

    const int qt = blockIdx.x * 4 + wave;  // 0..127
    const int h  = blockIdx.y;
    const int b  = blockIdx.z;
    const int bh = b * NUM_HEADS + h;
    const int q0 = qt * 16;

    const _Float16* Qb = Q  + bh * SEQ * HEAD_DIM;
    const _Float16* Kb = K  + bh * SEQ * HEAD_DIM;
    const _Float16* Vb = VT + bh * HEAD_DIM * SEQ;

    // Q A-fragments for d=0..31 and d=32..63 (reused across all key steps)
    Frag16 aq0, aq1;
    {
        const _Float16* qrow = Qb + (q0 + lr) * HEAD_DIM;
        aq0.h[0] = *(const v8h*)(qrow + hi * 16);
        aq0.h[1] = *(const v8h*)(qrow + hi * 16 + 8);
        aq1.h[0] = *(const v8h*)(qrow + 32 + hi * 16);
        aq1.h[1] = *(const v8h*)(qrow + 40 + hi * 16);
    }

    v8f o0 = zero8(), o1 = zero8(), o2 = zero8(), o3 = zero8();
    float mrun[8], lrun[8];
#pragma unroll
    for (int v = 0; v < 8; ++v) { mrun[v] = -3.0e38f; lrun[v] = 0.0f; }

    const int jend = ((q0 + 16 + 31) >> 5) << 5;
    for (int j = 0; j < jend; j += 32) {
        // ---- scores: two 16x16 tiles over keys [j, j+16) and [j+16, j+32)
        v8f s1, s2;
        {
            Frag16 bk;
            const _Float16* krow = Kb + (j + lr) * HEAD_DIM;   // key = j + lr
            bk.v = *(const v16h*)(krow + hi * 16);             // d = hi*16+0..15
            s1 = wmma16(aq0, bk, zero8());
            bk.v = *(const v16h*)(krow + 32 + hi * 16);        // d = 32+...
            s1 = wmma16(aq1, bk, s1);
            krow += 16 * HEAD_DIM;                             // key = j+16+lr
            bk.v = *(const v16h*)(krow + hi * 16);
            s2 = wmma16(aq0, bk, zero8());
            bk.v = *(const v16h*)(krow + 32 + hi * 16);
            s2 = wmma16(aq1, bk, s2);
        }

        // ---- online softmax over the 32 new keys (rows striped: M = v + 8*hi)
#pragma unroll
        for (int v = 0; v < 8; ++v) {
            const int rowq = q0 + v + 8 * hi;
            float a  = ((j + lr)      <= rowq) ? s1[v] : -1.0e30f;
            float bb = ((j + 16 + lr) <= rowq) ? s2[v] : -1.0e30f;
            float mx = fmaxf(a, bb);
#pragma unroll
            for (int off = 1; off < 16; off <<= 1)
                mx = fmaxf(mx, __shfl_xor(mx, off, 32));
            const float mnew  = fmaxf(mrun[v], mx);
            const float alpha = __expf(mrun[v] - mnew);
            mrun[v] = mnew;
            const float p1 = __expf(a - mnew);
            const float p2 = __expf(bb - mnew);
            float rs = p1 + p2;
#pragma unroll
            for (int off = 1; off < 16; off <<= 1)
                rs += __shfl_xor(rs, off, 32);
            lrun[v] = lrun[v] * alpha + rs;
            o0[v] *= alpha; o1[v] *= alpha; o2[v] *= alpha; o3[v] *= alpha;
            lds[wave][v + 8 * hi][lr]      = (_Float16)p1;
            lds[wave][v + 8 * hi][16 + lr] = (_Float16)p2;
        }
        asm volatile("" ::: "memory");  // keep ds stores before ds loads (same wave: LDS in-order)

        // ---- C-layout -> A-layout via LDS: P is a 16x32 f16 A fragment
        Frag16 ap;
        ap.h[0] = *(const v8h*)&lds[wave][lr][hi * 16];
        ap.h[1] = *(const v8h*)&lds[wave][lr][hi * 16 + 8];

        // ---- O += P * V   (V^T rows are contiguous in key -> one 32B load/frag)
        {
            Frag16 bv;
            const _Float16* vrow = Vb + lr * SEQ + j;  // d = dch*16 + lr
            bv.v = *(const v16h*)(vrow + hi * 16);
            o0 = wmma16(ap, bv, o0);
            vrow += 16 * SEQ;
            bv.v = *(const v16h*)(vrow + hi * 16);
            o1 = wmma16(ap, bv, o1);
            vrow += 16 * SEQ;
            bv.v = *(const v16h*)(vrow + hi * 16);
            o2 = wmma16(ap, bv, o2);
            vrow += 16 * SEQ;
            bv.v = *(const v16h*)(vrow + hi * 16);
            o3 = wmma16(ap, bv, o3);
        }
    }

    // ---- normalize and write attention output in [B,S,D] layout (f16)
    float linv[8];
#pragma unroll
    for (int v = 0; v < 8; ++v) linv[v] = 1.0f / lrun[v]; // >=1 (diagonal term)
    _Float16* out = AO + (b * SEQ + q0) * D_MODEL + h * HEAD_DIM + lr;
#pragma unroll
    for (int v = 0; v < 8; ++v) {
        _Float16* p = out + (v + 8 * hi) * D_MODEL;
        p[0]  = (_Float16)(o0[v] * linv[v]);
        p[16] = (_Float16)(o1[v] * linv[v]);
        p[32] = (_Float16)(o2[v] * linv[v]);
        p[48] = (_Float16)(o3[v] * linv[v]);
    }
}

// ---------------------------------------------------------------------------
extern "C" void kernel_launch(void* const* d_in, const int* in_sizes, int n_in,
                              void* d_out, int out_size, void* d_ws, size_t ws_size,
                              hipStream_t stream) {
    const float* x  = (const float*)d_in[0];
    const float* wq = (const float*)d_in[1];
    const float* wk = (const float*)d_in[2];
    const float* wv = (const float*)d_in[3];
    const float* wo = (const float*)d_in[4];

    char* ws = (char*)d_ws;
    const size_t MB = 1u << 20;
    _Float16* xh  = (_Float16*)(ws + 0 * MB);   // 16 MB
    _Float16* wqh = (_Float16*)(ws + 16 * MB);  //  2 MB
    _Float16* wkh = (_Float16*)(ws + 18 * MB);  //  2 MB
    _Float16* wvh = (_Float16*)(ws + 20 * MB);  //  2 MB
    _Float16* woh = (_Float16*)(ws + 22 * MB);  //  2 MB
    _Float16* Qh  = (_Float16*)(ws + 24 * MB);  // 16 MB
    _Float16* Kh  = (_Float16*)(ws + 40 * MB);  // 16 MB
    _Float16* VTh = (_Float16*)(ws + 56 * MB);  // 16 MB
    _Float16* AOh = (_Float16*)(ws + 72 * MB);  // 16 MB  (total 88 MB)

    const int n4x = TOKENS * D_MODEL / 4;
    cvt_f32_f16<<<(n4x + 255) / 256, 256, 0, stream>>>(x, xh, n4x);
    const int n4w = D_MODEL * D_MODEL / 4;
    cvt_f32_f16<<<(n4w + 255) / 256, 256, 0, stream>>>(wq, wqh, n4w);
    cvt_f32_f16<<<(n4w + 255) / 256, 256, 0, stream>>>(wk, wkh, n4w);
    cvt_f32_f16<<<(n4w + 255) / 256, 256, 0, stream>>>(wv, wvh, n4w);
    cvt_f32_f16<<<(n4w + 255) / 256, 256, 0, stream>>>(wo, woh, n4w);

    // block = 128 rows x 64 cols (8 waves, 32x32 tile each); x{q,k,v}
    qkv_kernel<<<dim3(TOKENS / 128, D_MODEL / 64, 3), 256, 0, stream>>>(
        xh, wqh, wkh, wvh, Qh, Kh, VTh);

    // 128 q-tiles x 16 heads x 4 batches; 4 waves (q-tiles) per block
    attn_kernel<<<dim3(32, NUM_HEADS, BATCH), 128, 0, stream>>>(Qh, Kh, VTh, AOh);

    oproj_kernel<<<dim3(TOKENS / 128, D_MODEL / 64), 256, 0, stream>>>(
        AOh, woh, (float*)d_out);
}
